// LSTMEncoder_58128087384343
// MI455X (gfx1250) — matrix operand order repaired and checked by reading, not compiled, not run on hardware
//
#include <hip/hip_runtime.h>

// ---------------------------------------------------------------------------
// Seq2seq LSTM (H=20, T=187) for MI455X / gfx1250.
// One wave32 owns 16 batch rows for the whole recurrence.
// Per step: gates[16x80] = h[16x20] @ Whh.T via 5x V_WMMA_F32_16X16X32_F16
// (K=20 zero-padded to 32), rank-1 input term + bias seeded into the C
// accumulator, activations via hardware v_tanh_f32, state update via an LDS
// layout switch. All sync is wave-local (LDS ops in-order per wave).
// ---------------------------------------------------------------------------

typedef __attribute__((ext_vector_type(16))) _Float16 v16h;
typedef __attribute__((ext_vector_type(8)))  _Float16 v8h;
typedef __attribute__((ext_vector_type(8)))  float    v8f;

#define HS     20      // hidden size
#define TS     187     // time steps (enc and dec)
#define NGATE  80      // 4*HS
#define NTILE  5       // 80 / 16
#define ROWS   16      // batch rows per wave (WMMA M)
#define WAVES  8       // waves per workgroup
#define KPAD   32      // K padded to WMMA K

// Fast activations: gfx1250 has V_TANH_F32; sigmoid(x) = 0.5*tanh(x/2)+0.5.
// Fallback avoids IEEE division (v_div_scale/fixup chains) via v_rcp_f32.
__device__ __forceinline__ float ftanh(float x) {
#if __has_builtin(__builtin_amdgcn_tanhf)
    return __builtin_amdgcn_tanhf(x);
#else
    x = fminf(fmaxf(x, -15.0f), 15.0f);
    float e = __expf(2.0f * x);
    return 1.0f - 2.0f * __builtin_amdgcn_rcpf(e + 1.0f);
#endif
}
__device__ __forceinline__ float fsig(float x) {
    return fmaf(ftanh(0.5f * x), 0.5f, 0.5f);
}

// B-fragment: B(k,n) = W[n][k], n = 16*t + (lane&15), half p -> K = p + 16*hi,
// zero-padded for K >= 20. Uses contiguous K order, same as the A fragment (K across
// halves, lanes 0-15 K=0..15, lanes 16-31 K=16..31, N = lane%16).
__device__ __forceinline__ v16h load_bfrag(const float* __restrict__ W,
                                           int t, int ln, int hi) {
    union { v16h v; _Float16 e[16]; } u;
    int n = t * 16 + ln;
#pragma unroll
    for (int p = 0; p < 16; ++p) {
        int k = p + hi * 16;
        u.e[p] = (k < HS) ? (_Float16)W[n * HS + k] : (_Float16)0.0f;
    }
    return u.v;
}

// One LSTM cell step for this wave's 16 rows. Wave-local: no barriers needed
// (LDS ops from the same wave complete in order).
__device__ __forceinline__ void lstm_step(
    int ln, int hi,
    const v16h* __restrict__ wf, const float* __restrict__ biasl,
    const float* __restrict__ wihl,
    _Float16 (*__restrict__ hb)[KPAD],  // [16][32] f16 h state (K-padded, zeros)
    float (*__restrict__ gb)[NGATE],    // [16][80] gate scratch
    float (*__restrict__ cb)[HS],       // [16][20] f32 c state
    float* __restrict__ xb,             // [16] per-row scalar input
    const float* __restrict__ wlb,      // [20] Wl (decoder only)
    float* __restrict__ ob,             // [16] dot accum (decoder only)
    bool write_he, bool dec) {

    // ---- A fragment from f16 h buffer (two ds_load_b128 per lane) ----
    // hi=0: halves 0-7 <- K0-7 (chunk0), halves 8-15 <- K8-15 (chunk1)
    // hi=1: halves 0-7 <- K16-23 (chunk2), halves 8-15 <- K24-31 (chunk3)
    const v8h* hp = (const v8h*)(&hb[ln][0]);
    union { v16h v; v8h h8[2]; } a;
    a.h8[0] = hp[2 * hi];
    a.h8[1] = hp[2 * hi + 1];

    // ---- per-row scalar input x (rank-1 term: gate[m][n] += Wih[n]*x[m]) ----
    float xr[8];
#pragma unroll
    for (int r = 0; r < 8; ++r) xr[r] = xb[r + hi * 8];

    // ---- 5 N-tiles: acc = bias + Wih*x, then one WMMA each, then act ----
#pragma unroll
    for (int t = 0; t < NTILE; ++t) {
        v8f acc;
#pragma unroll
        for (int r = 0; r < 8; ++r) acc[r] = fmaf(wihl[t], xr[r], biasl[t]);
        acc = __builtin_amdgcn_wmma_f32_16x16x32_f16(
            false, a.v, false, wf[t], (short)0, acc, false, false);
        int n = t * 16 + ln;                        // gate index for this lane
        bool is_g = (n >= 2 * HS) && (n < 3 * HS);  // g-gate -> tanh
#pragma unroll
        for (int r = 0; r < 8; ++r) {
            float v  = acc[r];
            // sigmoid(v) = 0.5*tanh(0.5v)+0.5 ; g-gate is plain tanh(v)
            float tv = ftanh(is_g ? v : 0.5f * v);
            float av = is_g ? tv : fmaf(tv, 0.5f, 0.5f);
            gb[r + hi * 8][n] = av;                 // D layout: M = r+8*hi, N = ln
        }
    }

    // ---- layout switch: lane -> (row m, hidden j); update c, h ----
#pragma unroll
    for (int it = 0; it < 10; ++it) {        // 16 rows * 20 hidden / 32 lanes
        int j = 2 * it + hi;
        int m = ln;
        float ig = gb[m][j];
        float fg = gb[m][HS + j];
        float gg = gb[m][2 * HS + j];
        float og = gb[m][3 * HS + j];
        float c2 = fmaf(fg, cb[m][j], ig * gg);
        float h2 = og * ftanh(c2);
        cb[m][j] = c2;
        hb[m][j] = (_Float16)h2;             // padding K>=20 stays zero
        if (write_he) gb[m][j] = h2;         // stash f32 he (i-gate slot, consumed)
        if (dec) atomicAdd(&ob[m], h2 * wlb[j]);  // h . Wl partial
    }
}

__global__ __launch_bounds__(256) void lstm_s2s_kernel(
    const float* __restrict__ inp,
    const float* __restrict__ Wih1, const float* __restrict__ Whh1,
    const float* __restrict__ bih1, const float* __restrict__ bhh1,
    const float* __restrict__ Wih2, const float* __restrict__ Whh2,
    const float* __restrict__ bih2, const float* __restrict__ bhh2,
    const float* __restrict__ Wl,   const float* __restrict__ bl,
    float* __restrict__ out, int Bn) {

    __shared__ _Float16 hbuf[WAVES][ROWS][KPAD] __attribute__((aligned(16)));
    __shared__ float    gbuf[WAVES][ROWS][NGATE];
    __shared__ float    cbuf[WAVES][ROWS][HS];
    __shared__ float    xbuf[WAVES][ROWS];
    __shared__ float    obuf[WAVES][ROWS];
    __shared__ float    wlb[HS];

    const int lane = threadIdx.x & 31;
    const int w    = threadIdx.x >> 5;
    const int hi   = lane >> 4;
    const int ln   = lane & 15;
    const int r0   = ((int)blockIdx.x * WAVES + w) * ROWS;

    // ---- init wave-local LDS state ----
    {
        _Float16* hp = &hbuf[w][0][0];
        for (int i = lane; i < ROWS * KPAD; i += 32) hp[i] = (_Float16)0.0f;
        float* cp = &cbuf[w][0][0];
        for (int i = lane; i < ROWS * HS; i += 32) cp[i] = 0.0f;
        xbuf[w][ln] = 0.0f;          // both lane halves write same value: benign
        obuf[w][ln] = 0.0f;
    }
    if (threadIdx.x < HS) wlb[threadIdx.x] = Wl[threadIdx.x];
    __syncthreads();                 // wlb visible to all waves (only barrier)

    const float blv = bl[0];

    // ---- encoder weights into registers ----
    v16h wf[NTILE]; float biasl[NTILE], wihl[NTILE];
#pragma unroll
    for (int t = 0; t < NTILE; ++t) {
        wf[t] = load_bfrag(Whh1, t, ln, hi);
        int n = t * 16 + ln;
        biasl[t] = bih1[n] + bhh1[n];
        wihl[t]  = Wih1[n];
    }

    // ---- encoder: 187 sequential steps ----
    for (int t = 0; t < TS; ++t) {
        // both lane halves load/store identical values -> no exec juggling
        float xv = inp[(size_t)(r0 + ln) * TS + t];
        xbuf[w][ln] = xv;
        lstm_step(ln, hi, wf, biasl, wihl,
                  hbuf[w], gbuf[w], cbuf[w], xbuf[w], wlb, obuf[w],
                  /*write_he=*/(t == TS - 1), /*dec=*/false);
    }

    // ---- log_softmax(he) from f32 snapshot in gbuf; reset decoder input ----
    if (lane < ROWS) {
        int m = lane;
        float mx = -1e30f;
        for (int j = 0; j < HS; ++j) mx = fmaxf(mx, gbuf[w][m][j]);
        float s = 0.0f;
        for (int j = 0; j < HS; ++j) s += __expf(gbuf[w][m][j] - mx);
        float lse = mx + __logf(s);
        if (r0 + m < Bn)
            for (int j = 0; j < HS; ++j)
                out[(size_t)Bn * TS + (size_t)(r0 + m) * HS + j] =
                    gbuf[w][m][j] - lse;
        xbuf[w][lane] = 0.0f;        // o0 = 0
        obuf[w][lane] = 0.0f;
    }

    // ---- decoder weights (reuse registers) ----
#pragma unroll
    for (int t = 0; t < NTILE; ++t) {
        wf[t] = load_bfrag(Whh2, t, ln, hi);
        int n = t * 16 + ln;
        biasl[t] = bih2[n] + bhh2[n];
        wihl[t]  = Wih2[n];
    }

    // ---- decoder: 187 autoregressive steps, output flipped ----
    for (int t = 0; t < TS; ++t) {
        lstm_step(ln, hi, wf, biasl, wihl,
                  hbuf[w], gbuf[w], cbuf[w], xbuf[w], wlb, obuf[w],
                  /*write_he=*/false, /*dec=*/true);
        // both halves compute o; only lanes<16 touch global memory
        float d = obuf[w][ln];       // LDS in-order: atomics above completed
        obuf[w][ln] = 0.0f;
        float o = fsig(d + blv);
        xbuf[w][ln] = o;             // next step's input (dup same-value write)
        if (lane < ROWS && r0 + lane < Bn)
            out[(size_t)(r0 + lane) * TS + (TS - 1 - t)] = o;
    }
}

extern "C" void kernel_launch(void* const* d_in, const int* in_sizes, int n_in,
                              void* d_out, int out_size, void* d_ws, size_t ws_size,
                              hipStream_t stream) {
    (void)n_in; (void)out_size; (void)d_ws; (void)ws_size;
    const float* inp  = (const float*)d_in[0];
    const float* Wih1 = (const float*)d_in[1];
    const float* Whh1 = (const float*)d_in[2];
    const float* bih1 = (const float*)d_in[3];
    const float* bhh1 = (const float*)d_in[4];
    const float* Wih2 = (const float*)d_in[5];
    const float* Whh2 = (const float*)d_in[6];
    const float* bih2 = (const float*)d_in[7];
    const float* bhh2 = (const float*)d_in[8];
    const float* Wl   = (const float*)d_in[9];
    const float* bl   = (const float*)d_in[10];
    float* out = (float*)d_out;

    int Bn = in_sizes[0] / TS;                 // 32768
    int rows_per_wg = WAVES * ROWS;            // 128
    int grid = (Bn + rows_per_wg - 1) / rows_per_wg;

    lstm_s2s_kernel<<<grid, WAVES * 32, 0, stream>>>(
        inp, Wih1, Whh1, bih1, bhh1, Wih2, Whh2, bih2, bhh2, Wl, bl, out, Bn);
}
